// Co_TransformerBlock_89240830476701
// MI455X (gfx1250) — matrix operand order repaired and checked by reading, not compiled, not run on hardware
//
#include <hip/hip_runtime.h>
#include <math.h>

// ---------------------------------------------------------------------------
// Types for CDNA5 WMMA (wave32): D(16x16 f32) = A(16x32 bf16) * B(32x16 bf16) + C
// ---------------------------------------------------------------------------
typedef __attribute__((ext_vector_type(16))) __bf16 v16bf;
typedef __attribute__((ext_vector_type(8)))  __bf16 v8bf;
typedef __attribute__((ext_vector_type(8)))  float  v8f;

#define DIMC   192
#define HEADS  8
#define HDIM   24            // DIM / HEADS
#define HID    510
#define HID2   1020          // 2*HID
#define BATCH  4
#define HWPIX  16384         // 128*128
#define EPSLN  1e-5f

static __device__ __forceinline__ v8f wmma_bf16(const v16bf& a, const v16bf& b, const v8f& c) {
    return __builtin_amdgcn_wmma_f32_16x16x32_bf16(false, a, false, b, (short)0, c, false, false);
}

union PackA { v16bf v; v8bf h[2]; };

// ---------------------------------------------------------------------------
// Weight convert f32[O,Kin] -> bf16[O,Kout] (zero pad K tail)
// ---------------------------------------------------------------------------
__global__ __launch_bounds__(256)
void wconv_kernel(const float* __restrict__ src, __bf16* __restrict__ dst,
                  int O, int Kin, int Kout)
{
    int i = blockIdx.x * blockDim.x + threadIdx.x;
    if (i >= O * Kout) return;
    int o = i / Kout, k = i - o * Kout;
    float v = (k < Kin) ? src[o * Kin + k] : 0.0f;
    dst[i] = (__bf16)v;
}

// ---------------------------------------------------------------------------
// LayerNorm over channels: in NCHW f32, out NHWC bf16
// ---------------------------------------------------------------------------
__global__ __launch_bounds__(256)
void ln_kernel(const float* __restrict__ x, const float* __restrict__ w,
               const float* __restrict__ b, __bf16* __restrict__ y)
{
    int idx = blockIdx.x * blockDim.x + threadIdx.x;         // global pixel
    int bb  = idx >> 14;
    int n   = idx & (HWPIX - 1);
    const float* xp = x + (size_t)bb * DIMC * HWPIX + n;
    float s = 0.f, s2 = 0.f;
    for (int c = 0; c < DIMC; ++c) {
        float v = xp[(size_t)c * HWPIX];
        s += v; s2 += v * v;
    }
    float mu  = s * (1.0f / DIMC);
    float var = s2 * (1.0f / DIMC) - mu * mu;
    float inv = rsqrtf(var + EPSLN);
    __bf16* yp = y + (size_t)idx * DIMC;
    for (int c = 0; c < DIMC; ++c) {
        float v = (xp[(size_t)c * HWPIX] - mu) * inv * w[c] + b[c];
        yp[c] = (__bf16)v;
    }
}

// ---------------------------------------------------------------------------
// GEMM: Y[pix, o] = sum_k W[o,k] * X[pix,k]   (NHWC bf16 in, NHWC bf16 out)
// 2x2 register-blocked: each wave computes a 32(o) x 32(pix) block
// (4 WMMAs per 8 b128 loads/lane -> 16 FLOP/B). blockDim=128 (4 waves).
// ---------------------------------------------------------------------------
__global__ __launch_bounds__(128)
void gemm_nhwc_bf16(const __bf16* __restrict__ X, int ldx,
                    const __bf16* __restrict__ W, int ldw, int K,
                    __bf16* __restrict__ Y, int ldy, int O)
{
    int wave = threadIdx.x >> 5, lane = threadIdx.x & 31;
    int m = lane & 15, hi = lane >> 4;
    int o0 = blockIdx.x * 32;
    int p0 = (blockIdx.y * 4 + wave) * 32;
    int ar0 = o0 + m;       if (ar0 >= O) ar0 = O - 1;
    int ar1 = o0 + 16 + m;  if (ar1 >= O) ar1 = O - 1;
    const __bf16* wp0 = W + (size_t)ar0 * ldw + hi * 8;
    const __bf16* wp1 = W + (size_t)ar1 * ldw + hi * 8;
    const __bf16* xp0 = X + (size_t)(p0 + m) * ldx + hi * 8;
    const __bf16* xp1 = X + (size_t)(p0 + 16 + m) * ldx + hi * 8;
    v8f a00 = {}, a01 = {}, a10 = {}, a11 = {};
    for (int kk = 0; kk < K; kk += 32) {
        __builtin_prefetch(xp0 + kk + 256, 0, 1);
        PackA A0, A1, B0, B1;
        A0.h[0] = *(const v8bf*)(wp0 + kk);
        A0.h[1] = *(const v8bf*)(wp0 + kk + 16);
        A1.h[0] = *(const v8bf*)(wp1 + kk);
        A1.h[1] = *(const v8bf*)(wp1 + kk + 16);
        B0.h[0] = *(const v8bf*)(xp0 + kk);
        B0.h[1] = *(const v8bf*)(xp0 + kk + 16);
        B1.h[0] = *(const v8bf*)(xp1 + kk);
        B1.h[1] = *(const v8bf*)(xp1 + kk + 16);
        a00 = wmma_bf16(A0.v, B0.v, a00);
        a01 = wmma_bf16(A0.v, B1.v, a01);
        a10 = wmma_bf16(A1.v, B0.v, a10);
        a11 = wmma_bf16(A1.v, B1.v, a11);
    }
    auto store_tile = [&](const v8f& acc, int obase, int pbase) {
        int ob = obase + 8 * hi;
        __bf16* yp = Y + (size_t)(pbase + m) * ldy + ob;
        if (ob + 8 <= O) {
            v8bf s;
            for (int r = 0; r < 8; ++r) s[r] = (__bf16)acc[r];
            *(v8bf*)yp = s;
        } else {
            for (int r = 0; r < 8; ++r) if (ob + r < O) yp[r] = (__bf16)acc[r];
        }
    };
    store_tile(a00, o0,      p0);
    store_tile(a01, o0,      p0 + 16);
    store_tile(a10, o0 + 16, p0);
    store_tile(a11, o0 + 16, p0 + 16);
}

// Same 2x2-blocked GEMM but epilogue: out NCHW f32 = acc + residual (NCHW f32)
__global__ __launch_bounds__(128)
void gemm_nchw_res(const __bf16* __restrict__ X, int ldx,
                   const __bf16* __restrict__ W, int ldw, int K,
                   const float* __restrict__ res, float* __restrict__ out, int O)
{
    int wave = threadIdx.x >> 5, lane = threadIdx.x & 31;
    int m = lane & 15, hi = lane >> 4;
    int o0 = blockIdx.x * 32;
    int p0 = (blockIdx.y * 4 + wave) * 32;
    int ar0 = o0 + m;       if (ar0 >= O) ar0 = O - 1;
    int ar1 = o0 + 16 + m;  if (ar1 >= O) ar1 = O - 1;
    const __bf16* wp0 = W + (size_t)ar0 * ldw + hi * 8;
    const __bf16* wp1 = W + (size_t)ar1 * ldw + hi * 8;
    const __bf16* xp0 = X + (size_t)(p0 + m) * ldx + hi * 8;
    const __bf16* xp1 = X + (size_t)(p0 + 16 + m) * ldx + hi * 8;
    v8f a00 = {}, a01 = {}, a10 = {}, a11 = {};
    for (int kk = 0; kk < K; kk += 32) {
        __builtin_prefetch(xp0 + kk + 256, 0, 1);
        PackA A0, A1, B0, B1;
        A0.h[0] = *(const v8bf*)(wp0 + kk);
        A0.h[1] = *(const v8bf*)(wp0 + kk + 16);
        A1.h[0] = *(const v8bf*)(wp1 + kk);
        A1.h[1] = *(const v8bf*)(wp1 + kk + 16);
        B0.h[0] = *(const v8bf*)(xp0 + kk);
        B0.h[1] = *(const v8bf*)(xp0 + kk + 16);
        B1.h[0] = *(const v8bf*)(xp1 + kk);
        B1.h[1] = *(const v8bf*)(xp1 + kk + 16);
        a00 = wmma_bf16(A0.v, B0.v, a00);
        a01 = wmma_bf16(A0.v, B1.v, a01);
        a10 = wmma_bf16(A1.v, B0.v, a10);
        a11 = wmma_bf16(A1.v, B1.v, a11);
    }
    auto store_tile = [&](const v8f& acc, int obase, int pbase) {
        int pix = pbase + m;
        int bb = pix >> 14, n = pix & (HWPIX - 1);
        size_t base = (size_t)bb * O * HWPIX + n;
        for (int r = 0; r < 8; ++r) {
            int o = obase + 8 * hi + r;
            if (o < O) {
                size_t idx = base + (size_t)o * HWPIX;
                out[idx] = acc[r] + res[idx];
            }
        }
    };
    store_tile(a00, o0,      p0);
    store_tile(a01, o0,      p0 + 16);
    store_tile(a10, o0 + 16, p0);
    store_tile(a11, o0 + 16, p0 + 16);
}

// ---------------------------------------------------------------------------
// Depthwise 3x3 (cross-correlation, SAME pad). Variants by output layout.
// ---------------------------------------------------------------------------
__global__ __launch_bounds__(256)
void dwconv_q_kernel(const __bf16* __restrict__ in, const float* __restrict__ w,
                     __bf16* __restrict__ outCHW)
{
    int t = blockIdx.x * blockDim.x + threadIdx.x;
    int cg = t % 24, pix = t / 24;
    int bb = pix >> 14, n = pix & (HWPIX - 1);
    int y = n >> 7, x = n & 127;
    int c0 = cg * 8;
    float acc[8]; for (int r = 0; r < 8; ++r) acc[r] = 0.f;
    for (int dy = -1; dy <= 1; ++dy) {
        int yy = y + dy; if (yy < 0 || yy > 127) continue;
        for (int dx = -1; dx <= 1; ++dx) {
            int xx = x + dx; if (xx < 0 || xx > 127) continue;
            const __bf16* p = in + ((size_t)(bb * HWPIX + yy * 128 + xx)) * DIMC + c0;
            int wi = (dy + 1) * 3 + (dx + 1);
            for (int r = 0; r < 8; ++r) acc[r] += (float)p[r] * w[(c0 + r) * 9 + wi];
        }
    }
    __bf16* op = outCHW + ((size_t)bb * DIMC + c0) * HWPIX + n;
    for (int r = 0; r < 8; ++r) op[(size_t)r * HWPIX] = (__bf16)acc[r];
}

__global__ __launch_bounds__(256)
void dwconv_kv_kernel(const __bf16* __restrict__ in, const float* __restrict__ w,
                      __bf16* __restrict__ kCHW, __bf16* __restrict__ vNHWC)
{
    int t = blockIdx.x * blockDim.x + threadIdx.x;
    int cg = t % 48, pix = t / 48;
    int bb = pix >> 14, n = pix & (HWPIX - 1);
    int y = n >> 7, x = n & 127;
    int c0 = cg * 8;
    float acc[8]; for (int r = 0; r < 8; ++r) acc[r] = 0.f;
    for (int dy = -1; dy <= 1; ++dy) {
        int yy = y + dy; if (yy < 0 || yy > 127) continue;
        for (int dx = -1; dx <= 1; ++dx) {
            int xx = x + dx; if (xx < 0 || xx > 127) continue;
            const __bf16* p = in + ((size_t)(bb * HWPIX + yy * 128 + xx)) * (2 * DIMC) + c0;
            int wi = (dy + 1) * 3 + (dx + 1);
            for (int r = 0; r < 8; ++r) acc[r] += (float)p[r] * w[(c0 + r) * 9 + wi];
        }
    }
    if (c0 < DIMC) {                       // k-part -> CHW
        __bf16* op = kCHW + ((size_t)bb * DIMC + c0) * HWPIX + n;
        for (int r = 0; r < 8; ++r) op[(size_t)r * HWPIX] = (__bf16)acc[r];
    } else {                               // v-part -> NHWC
        __bf16* op = vNHWC + (size_t)pix * DIMC + (c0 - DIMC);
        for (int r = 0; r < 8; ++r) op[r] = (__bf16)acc[r];
    }
}

// FF path: dwconv on 1020 channels (NHWC ld 1024), fused split + GELU gate.
// Output NHWC ld 512, channels 510/511 zeroed (clean K-pad for Wfout GEMM).
__global__ __launch_bounds__(256)
void dwconv_ff_kernel(const __bf16* __restrict__ in, const float* __restrict__ w,
                      __bf16* __restrict__ out)
{
    int t = blockIdx.x * blockDim.x + threadIdx.x;
    int cg = t % 64, pix = t / 64;
    int bb = pix >> 14, n = pix & (HWPIX - 1);
    int y = n >> 7, x = n & 127;
    int c0 = cg * 8;
    float a1[8], a2[8];
    for (int r = 0; r < 8; ++r) { a1[r] = 0.f; a2[r] = 0.f; }
    for (int dy = -1; dy <= 1; ++dy) {
        int yy = y + dy; if (yy < 0 || yy > 127) continue;
        for (int dx = -1; dx <= 1; ++dx) {
            int xx = x + dx; if (xx < 0 || xx > 127) continue;
            const __bf16* p = in + ((size_t)(bb * HWPIX + yy * 128 + xx)) * 1024;
            int wi = (dy + 1) * 3 + (dx + 1);
            for (int r = 0; r < 8; ++r) {
                int c = c0 + r;
                if (c < HID) {
                    a1[r] += (float)p[c]       * w[c * 9 + wi];
                    a2[r] += (float)p[HID + c] * w[(HID + c) * 9 + wi];
                }
            }
        }
    }
    __bf16* op = out + (size_t)pix * 512 + c0;
    for (int r = 0; r < 8; ++r) {
        int c = c0 + r;
        float g = 0.f;
        if (c < HID) g = 0.5f * a1[r] * (1.0f + erff(a1[r] * 0.70710678118f)) * a2[r];
        op[r] = (__bf16)g;
    }
}

// ---------------------------------------------------------------------------
// Row L2 norms over n=16384 of a CHW bf16 buffer (grid = B*192 rows)
// ---------------------------------------------------------------------------
__global__ __launch_bounds__(256)
void rownorm_kernel(const __bf16* __restrict__ X, float* __restrict__ norms)
{
    int row = blockIdx.x;
    const __bf16* p = X + (size_t)row * HWPIX;
    float s = 0.f;
    for (int i = threadIdx.x; i < HWPIX; i += 256) {
        float v = (float)p[i];
        s += v * v;
    }
    __shared__ float red[256];
    red[threadIdx.x] = s;
    __syncthreads();
    for (int off = 128; off > 0; off >>= 1) {
        if (threadIdx.x < off) red[threadIdx.x] += red[threadIdx.x + off];
        __syncthreads();
    }
    if (threadIdx.x == 0) norms[row] = fmaxf(sqrtf(red[0]), 1e-12f);
}

// ---------------------------------------------------------------------------
// Gram: raw[c,d] = sum_n q[c,n]*k[d,n] (unnormalized). One WG per (b,h),
// waves 0..3 -> prev tiles, 4..7 -> next tiles. WMMA over n in steps of 32.
// ---------------------------------------------------------------------------
__global__ __launch_bounds__(256)
void gram_kernel(const __bf16* __restrict__ Q, const __bf16* __restrict__ Kp,
                 const __bf16* __restrict__ Kn, float* __restrict__ rawP,
                 float* __restrict__ rawN)
{
    int bh = blockIdx.x, bb = bh >> 3, h = bh & 7;
    int wave = threadIdx.x >> 5, lane = threadIdx.x & 31;
    int m = lane & 15, hi = lane >> 4;
    const __bf16* Kbuf = (wave < 4) ? Kp : Kn;
    float* raw = (wave < 4) ? rawP : rawN;
    int w4 = wave & 3;
    int c0 = (w4 >> 1) * 16, d0 = (w4 & 1) * 16;
    int cl = c0 + m; if (cl > 23) cl = 23;       // clamp inside head (garbage ignored)
    int dl = d0 + m; if (dl > 23) dl = 23;
    const __bf16* qrow = Q    + (size_t)(bb * DIMC + h * HDIM + cl) * HWPIX;
    const __bf16* krow = Kbuf + (size_t)(bb * DIMC + h * HDIM + dl) * HWPIX;
    v8f acc = {};
    for (int nn = 0; nn < HWPIX; nn += 32) {
        PackA A, Bv;
        A.h[0]  = *(const v8bf*)(qrow + nn + hi * 8);
        A.h[1]  = *(const v8bf*)(qrow + nn + 16 + hi * 8);
        Bv.h[0] = *(const v8bf*)(krow + nn + hi * 8);
        Bv.h[1] = *(const v8bf*)(krow + nn + 16 + hi * 8);
        acc = wmma_bf16(A.v, Bv.v, acc);
    }
    for (int r = 0; r < 8; ++r) {
        int c = c0 + r + 8 * hi, d = d0 + m;
        if (c < HDIM && d < HDIM) raw[(size_t)bh * (HDIM * HDIM) + c * HDIM + d] = acc[r];
    }
}

// ---------------------------------------------------------------------------
// Finalize: scale by 1/(|q||k|)*temperature, softmax, product, softmax.
// Writes zero-padded 32x32 bf16 attn_co per (b,h).
// ---------------------------------------------------------------------------
__global__ __launch_bounds__(32)
void attn_finalize_kernel(const float* __restrict__ rawP, const float* __restrict__ rawN,
                          const float* __restrict__ nq, const float* __restrict__ nkp,
                          const float* __restrict__ nkn, const float* __restrict__ temp,
                          __bf16* __restrict__ attn_co)
{
    int bh = blockIdx.x, bb = bh >> 3, h = bh & 7;
    int c = threadIdx.x;
    __bf16* out = attn_co + (size_t)bh * 1024 + (size_t)c * 32;
    if (c >= HDIM) {
        for (int d = 0; d < 32; ++d) out[d] = (__bf16)0.f;
        return;
    }
    float t = temp[h];
    float invq = 1.0f / nq[bb * DIMC + h * HDIM + c];
    const float* rp = rawP + (size_t)bh * (HDIM * HDIM) + c * HDIM;
    const float* rn = rawN + (size_t)bh * (HDIM * HDIM) + c * HDIM;
    float ap[HDIM], an[HDIM], z[HDIM];
    float mp = -3.0e38f, mn = -3.0e38f;
    for (int d = 0; d < HDIM; ++d) {
        float sp = rp[d] * invq * (1.0f / nkp[bb * DIMC + h * HDIM + d]) * t;
        float sn = rn[d] * invq * (1.0f / nkn[bb * DIMC + h * HDIM + d]) * t;
        ap[d] = sp; an[d] = sn;
        mp = fmaxf(mp, sp); mn = fmaxf(mn, sn);
    }
    float ssp = 0.f, ssn = 0.f;
    for (int d = 0; d < HDIM; ++d) {
        ap[d] = expf(ap[d] - mp); ssp += ap[d];
        an[d] = expf(an[d] - mn); ssn += an[d];
    }
    float mz = -3.0e38f;
    for (int d = 0; d < HDIM; ++d) {
        z[d] = (ap[d] / ssp) * (an[d] / ssn);
        mz = fmaxf(mz, z[d]);
    }
    float sz = 0.f;
    for (int d = 0; d < HDIM; ++d) { z[d] = expf(z[d] - mz); sz += z[d]; }
    for (int d = 0; d < HDIM; ++d) out[d] = (__bf16)(z[d] / sz);
    for (int d = HDIM; d < 32; ++d) out[d] = (__bf16)0.f;
}

// vsum = vp + vn (in place into vp), bf16 elementwise
__global__ __launch_bounds__(256)
void add_bf16_kernel(__bf16* __restrict__ a, const __bf16* __restrict__ b)
{
    int i = blockIdx.x * blockDim.x + threadIdx.x;
    a[i] = (__bf16)((float)a[i] + (float)b[i]);
}

// ---------------------------------------------------------------------------
// out[b,h*24+c,pix] (NHWC) = attn_co(c,d=32) @ vsum(d,pix).  Single K-step WMMA.
// ---------------------------------------------------------------------------
__global__ __launch_bounds__(128)
void attn_apply_kernel(const __bf16* __restrict__ attn, const __bf16* __restrict__ vsum,
                       __bf16* __restrict__ out)
{
    int bh = blockIdx.x, bb = bh >> 3, h = bh & 7;
    int wave = threadIdx.x >> 5, lane = threadIdx.x & 31;
    int m = lane & 15, hi = lane >> 4;
    int n0 = (blockIdx.y * 4 + wave) * 16;
    const __bf16* ab = attn + (size_t)bh * 1024;
    PackA A0, A1, Bv;
    A0.h[0] = *(const v8bf*)(ab + (size_t)m * 32 + hi * 8);
    A0.h[1] = *(const v8bf*)(ab + (size_t)m * 32 + 16 + hi * 8);
    A1.h[0] = *(const v8bf*)(ab + (size_t)(16 + m) * 32 + hi * 8);
    A1.h[1] = *(const v8bf*)(ab + (size_t)(16 + m) * 32 + 16 + hi * 8);
    const __bf16* bp = vsum + ((size_t)(bb * HWPIX + n0 + m)) * DIMC + h * HDIM + hi * 8;
    Bv.h[0] = *(const v8bf*)(bp);
    Bv.h[1] = *(const v8bf*)(bp + 16);
    v8f acc0 = {}, acc1 = {};
    acc0 = wmma_bf16(A0.v, Bv.v, acc0);
    acc1 = wmma_bf16(A1.v, Bv.v, acc1);
    __bf16* op = out + ((size_t)(bb * HWPIX + n0 + m)) * DIMC + h * HDIM;
    v8bf s;
    for (int r = 0; r < 8; ++r) s[r] = (__bf16)acc0[r];
    *(v8bf*)(op + 8 * hi) = s;                       // c = 0..15
    if (hi == 0) {                                   // c = 16..23 only
        for (int r = 0; r < 8; ++r) s[r] = (__bf16)acc1[r];
        *(v8bf*)(op + 16) = s;
    }
}

// ---------------------------------------------------------------------------
// Workspace layout (bytes, all 256-aligned)
// ---------------------------------------------------------------------------
static constexpr size_t SZ_LN  = (size_t)BATCH * HWPIX * DIMC * 2;      // 25,165,824
static constexpr size_t SZ_KV  = (size_t)BATCH * HWPIX * 2 * DIMC * 2;  // 50,331,648
static constexpr size_t OFF_WQ    = 0;
static constexpr size_t OFF_WKVP  = OFF_WQ   + (size_t)192 * 192 * 2;
static constexpr size_t OFF_WKVN  = OFF_WKVP + (size_t)384 * 192 * 2;
static constexpr size_t OFF_WPROJ = OFF_WKVN + (size_t)384 * 192 * 2;
static constexpr size_t OFF_WFIN  = OFF_WPROJ + (size_t)192 * 192 * 2;
static constexpr size_t OFF_WFOUT = OFF_WFIN + (size_t)1020 * 192 * 2;
static constexpr size_t OFF_LNC   = OFF_WFOUT + (size_t)192 * 512 * 2;  // also LN2
static constexpr size_t OFF_LNP   = OFF_LNC + SZ_LN;
static constexpr size_t OFF_LNN   = OFF_LNP + SZ_LN;
static constexpr size_t OFF_X1    = OFF_LNN + SZ_LN;
static constexpr size_t OFF_NQ    = OFF_X1 + (size_t)BATCH * DIMC * HWPIX * 4;
static constexpr size_t OFF_NKP   = OFF_NQ  + 768 * 4;
static constexpr size_t OFF_NKN   = OFF_NKP + 768 * 4;
static constexpr size_t OFF_RAWP  = OFF_NKN + 768 * 4;
static constexpr size_t OFF_RAWN  = OFF_RAWP + (size_t)32 * 576 * 4;
static constexpr size_t OFF_ACO   = OFF_RAWN + (size_t)32 * 576 * 4;
static constexpr size_t OFF_R     = OFF_ACO + (size_t)32 * 1024 * 2;
static constexpr size_t OFF_Q0    = OFF_R;                    // also out_attn, later FF1
static constexpr size_t OFF_KVP0  = OFF_Q0   + SZ_LN;
static constexpr size_t OFF_KVN0  = OFF_KVP0 + SZ_KV;
static constexpr size_t OFF_QC    = OFF_KVN0 + SZ_KV;
static constexpr size_t OFF_KPC   = OFF_QC  + SZ_LN;
static constexpr size_t OFF_KNC   = OFF_KPC + SZ_LN;
static constexpr size_t OFF_VP    = OFF_KNC + SZ_LN;          // + 256 B pad for tail read
static constexpr size_t OFF_VN    = OFF_VP + SZ_LN + 256;
static constexpr size_t OFF_FF1   = OFF_R;                            // 134,217,728
static constexpr size_t OFF_FFG   = OFF_R + (size_t)BATCH * HWPIX * 1024 * 2;
static constexpr size_t WS_NEED   = OFF_VN + SZ_LN;

extern "C" void kernel_launch(void* const* d_in, const int* in_sizes, int n_in,
                              void* d_out, int out_size, void* d_ws, size_t ws_size,
                              hipStream_t stream)
{
    (void)in_sizes; (void)n_in; (void)out_size;
    if (ws_size < WS_NEED) return;

    const float* x_curr = (const float*)d_in[0];
    const float* x_prev = (const float*)d_in[1];
    const float* x_next = (const float*)d_in[2];
    const float* ln1_w  = (const float*)d_in[3];
    const float* ln1_b  = (const float*)d_in[4];
    const float* ln2_w  = (const float*)d_in[5];
    const float* ln2_b  = (const float*)d_in[6];
    const float* temp   = (const float*)d_in[7];
    const float* Wq     = (const float*)d_in[8];
    const float* Wq_dw  = (const float*)d_in[9];
    const float* Wkvp   = (const float*)d_in[10];
    const float* Wkvp_dw= (const float*)d_in[11];
    const float* Wkvn   = (const float*)d_in[12];
    const float* Wkvn_dw= (const float*)d_in[13];
    const float* Wproj  = (const float*)d_in[14];
    const float* Wfin   = (const float*)d_in[15];
    const float* Wf_dw  = (const float*)d_in[16];
    const float* Wfout  = (const float*)d_in[17];
    float* out = (float*)d_out;
    char*  ws  = (char*)d_ws;

    auto B16 = [&](size_t off) { return (__bf16*)(ws + off); };
    auto F32 = [&](size_t off) { return (float*)(ws + off); };

    // 0) weights -> bf16 (Wfout zero-padded K 510 -> 512)
    wconv_kernel<<<144, 256, 0, stream>>>(Wq,    B16(OFF_WQ),    192, 192, 192);
    wconv_kernel<<<288, 256, 0, stream>>>(Wkvp,  B16(OFF_WKVP),  384, 192, 192);
    wconv_kernel<<<288, 256, 0, stream>>>(Wkvn,  B16(OFF_WKVN),  384, 192, 192);
    wconv_kernel<<<144, 256, 0, stream>>>(Wproj, B16(OFF_WPROJ), 192, 192, 192);
    wconv_kernel<<<765, 256, 0, stream>>>(Wfin,  B16(OFF_WFIN), 1020, 192, 192);
    wconv_kernel<<<384, 256, 0, stream>>>(Wfout, B16(OFF_WFOUT), 192, 510, 512);

    // 1) LayerNorm1 on all three frames (NCHW f32 -> NHWC bf16)
    ln_kernel<<<256, 256, 0, stream>>>(x_curr, ln1_w, ln1_b, B16(OFF_LNC));
    ln_kernel<<<256, 256, 0, stream>>>(x_prev, ln1_w, ln1_b, B16(OFF_LNP));
    ln_kernel<<<256, 256, 0, stream>>>(x_next, ln1_w, ln1_b, B16(OFF_LNN));

    // 2) 1x1 convs via 2x2-blocked WMMA GEMM (o-blocks of 32, 128 pixels/block)
    gemm_nhwc_bf16<<<dim3( 6, 512), 128, 0, stream>>>(B16(OFF_LNC), 192, B16(OFF_WQ),   192, 192, B16(OFF_Q0),   192, 192);
    gemm_nhwc_bf16<<<dim3(12, 512), 128, 0, stream>>>(B16(OFF_LNP), 192, B16(OFF_WKVP), 192, 192, B16(OFF_KVP0), 384, 384);
    gemm_nhwc_bf16<<<dim3(12, 512), 128, 0, stream>>>(B16(OFF_LNN), 192, B16(OFF_WKVN), 192, 192, B16(OFF_KVN0), 384, 384);

    // 3) depthwise 3x3: q -> CHW ; k -> CHW, v -> NHWC
    dwconv_q_kernel <<<6144,  256, 0, stream>>>(B16(OFF_Q0),   Wq_dw,   B16(OFF_QC));
    dwconv_kv_kernel<<<12288, 256, 0, stream>>>(B16(OFF_KVP0), Wkvp_dw, B16(OFF_KPC), B16(OFF_VP));
    dwconv_kv_kernel<<<12288, 256, 0, stream>>>(B16(OFF_KVN0), Wkvn_dw, B16(OFF_KNC), B16(OFF_VN));

    // 4) L2 row norms (folded into attention scaling)
    rownorm_kernel<<<768, 256, 0, stream>>>(B16(OFF_QC),  F32(OFF_NQ));
    rownorm_kernel<<<768, 256, 0, stream>>>(B16(OFF_KPC), F32(OFF_NKP));
    rownorm_kernel<<<768, 256, 0, stream>>>(B16(OFF_KNC), F32(OFF_NKN));

    // 5) Gram matrices (WMMA over n), finalize softmaxes -> padded bf16 attn
    gram_kernel<<<32, 256, 0, stream>>>(B16(OFF_QC), B16(OFF_KPC), B16(OFF_KNC),
                                        F32(OFF_RAWP), F32(OFF_RAWN));
    attn_finalize_kernel<<<32, 32, 0, stream>>>(F32(OFF_RAWP), F32(OFF_RAWN),
                                                F32(OFF_NQ), F32(OFF_NKP), F32(OFF_NKN),
                                                temp, B16(OFF_ACO));

    // 6) vsum = v_p + v_n (in place), then out = attn_co @ vsum (WMMA)
    add_bf16_kernel<<<49152, 256, 0, stream>>>(B16(OFF_VP), B16(OFF_VN));
    attn_apply_kernel<<<dim3(32, 256), 128, 0, stream>>>(B16(OFF_ACO), B16(OFF_VP), B16(OFF_Q0));

    // 7) proj + residual -> x1 (NCHW f32)
    gemm_nchw_res<<<dim3(6, 512), 128, 0, stream>>>(B16(OFF_Q0), 192, B16(OFF_WPROJ), 192, 192,
                                                    x_curr, F32(OFF_X1), 192);

    // 8) FF: LN2 -> Wfin GEMM -> dwconv+GELU gate -> Wfout GEMM + residual -> d_out
    ln_kernel<<<256, 256, 0, stream>>>(F32(OFF_X1), ln2_w, ln2_b, B16(OFF_LNC));
    gemm_nhwc_bf16<<<dim3(32, 512), 128, 0, stream>>>(B16(OFF_LNC), 192, B16(OFF_WFIN), 192, 192,
                                                      B16(OFF_FF1), 1024, 1020);
    dwconv_ff_kernel<<<16384, 256, 0, stream>>>(B16(OFF_FF1), Wf_dw, B16(OFF_FFG));
    gemm_nchw_res<<<dim3(6, 512), 128, 0, stream>>>(B16(OFF_FFG), 512, B16(OFF_WFOUT), 512, 512,
                                                    F32(OFF_X1), out, 192);
}
